// DifferentiableSinkhorn_43868795961525
// MI455X (gfx1250) — compile-verified, matrix operand order checked by reading, not run on hardware
//
#include <hip/hip_runtime.h>

typedef __bf16 v16bf __attribute__((ext_vector_type(16)));
typedef __bf16 bf16x2 __attribute__((ext_vector_type(2)));
typedef float  v8f   __attribute__((ext_vector_type(8)));

#define BATCH 64
#define NN 1024
#define NM (1024 * 1024)

__device__ __forceinline__ unsigned short f32_to_bf16(float f) {
  union { float f; unsigned int u; } v; v.f = f;
  unsigned int r = v.u + 0x7FFFu + ((v.u >> 16) & 1u);   // round-to-nearest-even
  return (unsigned short)(r >> 16);
}
__device__ __forceinline__ float bf16_to_f32(unsigned int hs) {
  union { unsigned int u; float f; } v; v.u = hs << 16;
  return v.f;
}

union Frag8 {
  uint4 q[2];
  unsigned int u[8];
  bf16x2 h2[8];
  v16bf v;
};

// ---------------- prep: K[b,n,m] = bf16(exp(la*scale+bias)); vbf = 1.0 ----------------
__global__ void sinkhorn_prep(const float* __restrict__ la,
                              const float* __restrict__ scale,
                              const float* __restrict__ bias,
                              unsigned short* __restrict__ kbuf,
                              unsigned short* __restrict__ vbf) {
  int i4 = blockIdx.x * blockDim.x + threadIdx.x;        // one float4 per thread
  float s = scale[0], b0 = bias[0];
  float4 x = ((const float4*)la)[i4];
  unsigned int d0 = (unsigned int)f32_to_bf16(__expf(fmaf(x.x, s, b0)))
                  | ((unsigned int)f32_to_bf16(__expf(fmaf(x.y, s, b0))) << 16);
  unsigned int d1 = (unsigned int)f32_to_bf16(__expf(fmaf(x.z, s, b0)))
                  | ((unsigned int)f32_to_bf16(__expf(fmaf(x.w, s, b0))) << 16);
  ((uint2*)kbuf)[i4] = make_uint2(d0, d1);
  if (i4 < BATCH * NN) vbf[i4] = 0x3F80;                 // bf16 1.0
}

// ---------------- row pass: u[b,n] = 1 / sum_m K[b,n,m] * v[m]  (WMMA) ----------------
// Wave handles 16 rows. A = K-tile (16x32) premultiplied by v (bf16), B = ones.
// With B = all-ones, D[i,*] = rowsum(A[i,:]) regardless of the k-permutation inside the
// A fragment; we only rely on lanes L and L+16 mapping to row (L&15).
__global__ void sinkhorn_row(const unsigned short* __restrict__ kbuf,
                             const unsigned short* __restrict__ vbf,
                             float* __restrict__ ubuf) {
  const int lane = threadIdx.x & 31;
  const int wid  = threadIdx.x >> 5;                     // 8 waves
  const int b    = blockIdx.x >> 3;                      // 8 blocks per batch
  const int rb   = blockIdx.x & 7;
  const int r0   = rb * 128 + wid * 16;
  const int row  = r0 + (lane & 15);
  const int h    = lane >> 4;                            // which 16-wide k half

  const uint4* kp = (const uint4*)(kbuf + (size_t)b * NM + (size_t)row * NN);
  const uint4* vp = (const uint4*)(vbf + b * NN);

  Frag8 ones;
  #pragma unroll
  for (int t = 0; t < 8; ++t) ones.u[t] = 0x3F803F80u;   // bf16x2 {1,1}

  v8f acc = {};
  #pragma unroll 4
  for (int m0 = 0; m0 < NN; m0 += 32) {
    const int q = (m0 >> 3) + (h << 1);                  // 32B chunk: cols m0+16h .. +15
    Frag8 a, vv;
    a.q[0]  = kp[q];  a.q[1]  = kp[q + 1];
    vv.q[0] = vp[q];  vv.q[1] = vp[q + 1];
    #pragma unroll
    for (int t = 0; t < 8; ++t) a.h2[t] = a.h2[t] * vv.h2[t];   // v_pk_mul_bf16
    acc = __builtin_amdgcn_wmma_f32_16x16x32_bf16(false, a.v, false, ones.v,
                                                  (short)0, acc, false, false);
  }

  // D layout: VGPR r, lanes 0-15 -> M=r (N=lane); lanes 16-31 -> M=r+8.
  if ((lane & 15) == 0) {
    const int rbase = r0 + h * 8;
    #pragma unroll
    for (int t = 0; t < 8; ++t)
      ubuf[b * NN + rbase + t] = 1.0f / acc[t];
  }
}

// ---------------- col pass: v[b,m] = 1 / sum_n K[b,n,m] * u[n]  (VALU, deterministic) --
__global__ void sinkhorn_col(const unsigned short* __restrict__ kbuf,
                             const float* __restrict__ ubuf,
                             float* __restrict__ vbuf,
                             unsigned short* __restrict__ vbf) {
  __shared__ float part[8][64];
  const int lane = threadIdx.x & 31;
  const int wid  = threadIdx.x >> 5;                     // 8 waves split the rows
  const int b    = blockIdx.x >> 4;                      // 16 blocks per batch
  const int cb   = blockIdx.x & 15;
  const int c0   = cb * 64;                              // 64 columns per block

  const unsigned int* kp = (const unsigned int*)(kbuf + (size_t)b * NM);
  const float* up = ubuf + b * NN;
  const int dcol = (c0 >> 1) + lane;                     // dword = 2 bf16 columns

  float acc0 = 0.f, acc1 = 0.f;
  const int rbeg = wid * 128;
  #pragma unroll 4
  for (int r = rbeg; r < rbeg + 128; ++r) {
    unsigned int d = kp[(size_t)r * 512 + dcol];         // coalesced 128B per wave
    float uu = up[r];
    acc0 = fmaf(bf16_to_f32(d & 0xFFFFu), uu, acc0);
    acc1 = fmaf(bf16_to_f32(d >> 16),     uu, acc1);
  }
  part[wid][2 * lane]     = acc0;
  part[wid][2 * lane + 1] = acc1;
  __syncthreads();

  if (threadIdx.x < 64) {
    float s = 0.f;
    #pragma unroll
    for (int w = 0; w < 8; ++w) s += part[w][threadIdx.x];  // fixed-order reduce
    float v = 1.0f / s;
    vbuf[b * NN + c0 + threadIdx.x] = v;
    vbf[b * NN + c0 + threadIdx.x]  = f32_to_bf16(v);
  }
}

// ---------------- final: P = exp(la*s+b) * u[n] * v[m]  (full f32 from original input) -
__global__ void sinkhorn_final(const float* __restrict__ la,
                               const float* __restrict__ scale,
                               const float* __restrict__ bias,
                               const float* __restrict__ ubuf,
                               const float* __restrict__ vbuf,
                               float* __restrict__ out) {
  int i4 = blockIdx.x * blockDim.x + threadIdx.x;
  int b  = i4 >> 18;                                     // 2^18 float4 per batch
  int n  = (i4 >> 8) & 1023;                             // 256 float4 per row
  int mb = i4 & 255;
  float s = scale[0], b0 = bias[0];
  float4 x  = ((const float4*)la)[i4];
  float  uu = ubuf[b * NN + n];
  float4 vv = ((const float4*)(vbuf + b * NN))[mb];
  float4 r;
  r.x = __expf(fmaf(x.x, s, b0)) * uu * vv.x;
  r.y = __expf(fmaf(x.y, s, b0)) * uu * vv.y;
  r.z = __expf(fmaf(x.z, s, b0)) * uu * vv.z;
  r.w = __expf(fmaf(x.w, s, b0)) * uu * vv.w;
  ((float4*)out)[i4] = r;
}

extern "C" void kernel_launch(void* const* d_in, const int* in_sizes, int n_in,
                              void* d_out, int out_size, void* d_ws, size_t ws_size,
                              hipStream_t stream) {
  const float* la    = (const float*)d_in[0];
  const float* scale = (const float*)d_in[1];
  const float* bias  = (const float*)d_in[2];
  float* out = (float*)d_out;

  // bf16 K (128 MB, L2-resident) parked in the first half of d_out; it is dead
  // before sinkhorn_final overwrites d_out (final reads only la/u/v).
  unsigned short* kbuf = (unsigned short*)d_out;

  // Scratch vectors in d_ws: u (256KB) | v (256KB) | v_bf16 (128KB)
  float* ubuf = (float*)d_ws;
  float* vbuf = ubuf + BATCH * NN;
  unsigned short* vbf = (unsigned short*)(vbuf + BATCH * NN);

  const int prep_blocks = (BATCH * NM / 4) / 256;        // 65536
  sinkhorn_prep<<<prep_blocks, 256, 0, stream>>>(la, scale, bias, kbuf, vbf);

  for (int it = 0; it < 20; ++it) {
    sinkhorn_row<<<BATCH * 8, 256, 0, stream>>>(kbuf, vbf, ubuf);
    sinkhorn_col<<<BATCH * 16, 256, 0, stream>>>(kbuf, ubuf, vbuf, vbf);
  }

  sinkhorn_final<<<prep_blocks, 256, 0, stream>>>(la, scale, bias, ubuf, vbuf, out);
}
